// WeightOnlyInt4Linear_1726576857544
// MI455X (gfx1250) — compile-verified, matrix operand order checked by reading, not run on hardware
//
#include <hip/hip_runtime.h>

typedef __attribute__((ext_vector_type(16))) _Float16     v16h;
typedef __attribute__((ext_vector_type(2)))  _Float16     h2;
typedef __attribute__((ext_vector_type(8)))  float        v8f;
typedef __attribute__((ext_vector_type(4)))  float        f32x4;
typedef __attribute__((ext_vector_type(4)))  int          i32x4;
typedef __attribute__((ext_vector_type(4)))  unsigned int u32x4;

#define K_DIM    4096
#define N_OUT    4096
#define GROUP    128
#define NG       (K_DIM / GROUP)   // 32 quant groups
#define BM       128
#define BN       128
#define BK       32
#define NT       (K_DIM / BK)      // 128 k-iterations
#define LDAF     (BK + 4)   // A tile: f32, 36 DW stride = 144B (16B-aligned, conflict-free rotation)
#define LDBH     (BK + 8)   // B tile: f16, 40 halves = 80B stride (16B-aligned, conflict-free)
#define NTHREADS 256

// ---- gfx1250 async global->LDS copy (ASYNCcnt-tracked), with synchronous fallback ----
#if defined(__has_builtin)
#  if __has_builtin(__builtin_amdgcn_global_load_async_to_lds_b128)
#    define USE_ASYNC_LDS 1
#  endif
#endif
#ifndef USE_ASYNC_LDS
#  define USE_ASYNC_LDS 0
#endif

#if USE_ASYNC_LDS
// builtin signature: (int4 addrspace(1)*, int4 addrspace(3)*, imm offset, imm cpol)
typedef __attribute__((address_space(1))) i32x4 g_i32x4;
typedef __attribute__((address_space(3))) i32x4 l_i32x4;
#endif

static __device__ __forceinline__ void async_copy_b128(const float* gsrc, float* ldst) {
#if USE_ASYNC_LDS
    __builtin_amdgcn_global_load_async_to_lds_b128((g_i32x4*)gsrc, (l_i32x4*)ldst, 0, 0);
#else
    *(f32x4*)ldst = *(const f32x4*)gsrc;   // fallback: through VGPRs
#endif
}

static __device__ __forceinline__ void wait_async_all() {
#if USE_ASYNC_LDS
#  if defined(__has_builtin) && __has_builtin(__builtin_amdgcn_s_wait_asynccnt)
    __builtin_amdgcn_s_wait_asynccnt(0);
#  else
    asm volatile("s_wait_asynccnt 0" ::: "memory");
#  endif
#endif
}

__global__ __launch_bounds__(NTHREADS)
void wo_int4_wmma_gemm(const float* __restrict__ X,
                       const int*   __restrict__ Wq,
                       const float* __restrict__ SZ,
                       float* __restrict__ Out,
                       int M)
{
    __shared__ float    Af[2][BM * LDAF];   // activations, raw f32, [row][k]
    __shared__ _Float16 Bs[2][BN * LDBH];   // dequantized weights, f16, [n][k]

    const int tid   = threadIdx.x;
    const int lane  = tid & 31;
    const int wid   = tid >> 5;
    const int waveM = wid >> 1;        // 0..3 -> 32-row slab
    const int waveN = wid & 1;         // 0..1 -> 64-col slab
    const int r15   = lane & 15;
    const int hi    = lane >> 4;       // 0: lanes 0-15, 1: lanes 16-31

    const int blockM = blockIdx.y * BM;
    const int blockN = blockIdx.x * BN;
    if (blockM >= M) return;

    // B-loader mapping: each thread owns a half-row (16 K elements)
    const int lrow = tid >> 1;          // 0..127
    const int hoff = (tid & 1) * 16;    // 0 or 16 along K

    const float* xrow = X  + (size_t)(blockM + lrow) * K_DIM + hoff;
    const int*   wrow = Wq + (size_t)(blockN + lrow) * K_DIM + hoff;
    const float* szrow = SZ + (size_t)(blockN + lrow) * 2;    // + g*N_OUT*2 per group

    // ---- A tile: 128 rows x 32 f32 = 1024 x 16B chunks; 4 async b128 copies/thread ----
    auto load_A = [&](int buf, int k0) {
        #pragma unroll
        for (int j = 0; j < 4; ++j) {
            const int idx = tid + j * NTHREADS;     // 0..1023
            const int r   = idx >> 3;               // row 0..127
            const int c   = (idx & 7) << 2;         // f32 col 0,4,..,28
            async_copy_b128(X + (size_t)(blockM + r) * K_DIM + k0 + c,
                            &Af[buf][r * LDAF + c]);
        }
    };

    // ---- B tile: int4-in-int32 -> f16 via exponent-splice trick ----
    // 0x6400|q is f16(1024+q) exactly; (f16(1024+q) - 1032) is exact q-8;
    // w = fma(q-8, s, z) in packed f16 with single rounding.
    const h2 cm1032 = { (_Float16)(-1032.0f), (_Float16)(-1032.0f) };
    auto load_B = [&](int buf, int k0, int g) {
        const float2 sz = *(const float2*)(szrow + (size_t)g * N_OUT * 2);
        const h2 sv = { (_Float16)sz.x, (_Float16)sz.x };
        const h2 zv = { (_Float16)sz.y, (_Float16)sz.y };
        const int* wp = wrow + k0;
        _Float16*  bs = &Bs[buf][lrow * LDBH + hoff];
        #pragma unroll
        for (int j = 0; j < 2; ++j) {
            i32x4 q0 = *(const i32x4*)(wp + j * 8);
            i32x4 q1 = *(const i32x4*)(wp + j * 8 + 4);
            u32x4 hw;
            #pragma unroll
            for (int p = 0; p < 2; ++p) {
                unsigned u0 = (unsigned)q0[2*p] | ((unsigned)q0[2*p+1] << 16) | 0x64006400u;
                unsigned u1 = (unsigned)q1[2*p] | ((unsigned)q1[2*p+1] << 16) | 0x64006400u;
                h2 w0 = (__builtin_bit_cast(h2, u0) + cm1032) * sv + zv;  // v_pk_add + v_pk_fma
                h2 w1 = (__builtin_bit_cast(h2, u1) + cm1032) * sv + zv;
                hw[p]     = __builtin_bit_cast(unsigned, w0);
                hw[p + 2] = __builtin_bit_cast(unsigned, w1);
            }
            *(u32x4*)(bs + j * 8) = hw;                                   // ds_store_b128
        }
    };

    v8f acc[2][4] = {};

    load_A(0, 0);
    load_B(0, 0, 0);
    wait_async_all();
    __syncthreads();

    for (int gt = 0; gt < NG; ++gt) {          // one quant group = 4 BK sub-tiles
        #pragma unroll
        for (int i = 0; i < 4; ++i) {
            const int kt  = gt * 4 + i;
            const int cur = i & 1;             // compile-time buffer parity
            if (kt + 1 < NT) {                 // produce next sub-tile (imm k-offsets)
                const int koff = (i + 1) * BK;
                load_A(cur ^ 1, gt * GROUP + koff);
                load_B(cur ^ 1, gt * GROUP + koff, gt + (koff >> 7));
            }
            if (kt + 2 < NT) {                 // WGP-scope prefetch of k+2 tiles
                __builtin_prefetch(xrow + gt * GROUP + (i + 2) * BK, 0, 3);
                __builtin_prefetch(wrow + gt * GROUP + (i + 2) * BK, 0, 3);
            }

            const float*    aF = Af[cur];
            const _Float16* bs = Bs[cur];
            // per-lane K chunks per ISA 16-bit A layout:
            // lanes 0-15: K 0..7 then 16..23; lanes 16-31: +8
            const int kc0 = hi * 8;
            const int kc1 = kc0 + 16;

            v16h afrag[2];
            #pragma unroll
            for (int mt = 0; mt < 2; ++mt) {
                const float* p = aF + (waveM * 32 + mt * 16 + r15) * LDAF;
                f32x4 x0 = *(const f32x4*)(p + kc0);        // ds_load_b128 x4
                f32x4 x1 = *(const f32x4*)(p + kc0 + 4);
                f32x4 x2 = *(const f32x4*)(p + kc1);
                f32x4 x3 = *(const f32x4*)(p + kc1 + 4);
                v16h a;
                #pragma unroll
                for (int e = 0; e < 4; ++e) {               // v_cvt_pk_f16_f32 x8
                    a[e]      = (_Float16)x0[e];
                    a[e + 4]  = (_Float16)x1[e];
                    a[e + 8]  = (_Float16)x2[e];
                    a[e + 12] = (_Float16)x3[e];
                }
                afrag[mt] = a;
            }
            #pragma unroll
            for (int nt = 0; nt < 4; ++nt) {
                const _Float16* p = bs + (waveN * 64 + nt * 16 + r15) * LDBH;
                v16h bfrag;
                u32x4* d = (u32x4*)&bfrag;
                d[0] = *(const u32x4*)(p + kc0);            // ds_load_b128 x2
                d[1] = *(const u32x4*)(p + kc1);
                #pragma unroll
                for (int mt = 0; mt < 2; ++mt) {
                    acc[mt][nt] = __builtin_amdgcn_wmma_f32_16x16x32_f16(
                        /*neg_a=*/false, afrag[mt], /*neg_b=*/false, bfrag,
                        /*c_mod=*/(short)0, acc[mt][nt],
                        /*reuse_a=*/false, /*reuse_b=*/false);
                }
            }
            wait_async_all();       // our async LDS writes for buf cur^1 landed
            __syncthreads();        // publish to workgroup
        }
    }

    // ---- writeback: VGPR v -> row (v + 8*hi), lane%16 -> col ----
    float* outp = Out + (size_t)(blockM + waveM * 32 + hi * 8) * N_OUT
                      + blockN + waveN * 64 + r15;
    #pragma unroll
    for (int mt = 0; mt < 2; ++mt) {
        #pragma unroll
        for (int nt = 0; nt < 4; ++nt) {
            #pragma unroll
            for (int v = 0; v < 8; ++v) {
                outp[(size_t)(mt * 16 + v) * N_OUT + nt * 16] = acc[mt][nt][v];
            }
        }
    }
}

extern "C" void kernel_launch(void* const* d_in, const int* in_sizes, int n_in,
                              void* d_out, int out_size, void* d_ws, size_t ws_size,
                              hipStream_t stream) {
    const float* X   = (const float*)d_in[0];         // [M, 4096] f32
    const int*   Wq  = (const int*)d_in[1];           // [4096, 4096] int32 (values 0..15)
    const float* SZ  = (const float*)d_in[2];         // [32, 4096, 2] f32
    float*       Out = (float*)d_out;                 // [M, 4096] f32

    const int M = in_sizes[0] / K_DIM;                // 4*2048 = 8192
    dim3 grid(N_OUT / BN, (M + BM - 1) / BM);         // (32, 64)
    wo_int4_wmma_gemm<<<grid, NTHREADS, 0, stream>>>(X, Wq, SZ, Out, M);
}
